// GCN1000_20469814133395
// MI455X (gfx1250) — matrix operand order repaired and checked by reading, not compiled, not run on hardware
//
#include <hip/hip_runtime.h>
#include <math.h>

typedef __bf16 v16bf __attribute__((ext_vector_type(16)));
typedef __bf16 v4bf  __attribute__((ext_vector_type(4)));
typedef float  v8f   __attribute__((ext_vector_type(8)));

#define F_IN 512
#define HID  64
#define NCLS 16

__device__ __forceinline__ __bf16 f2bf(float f) { return (__bf16)f; }

// Fragment-layout helpers (ISA 7.12.2, 16-bit A 16x32):
//   lane = m + 16*hi, hi = (k>>3)&1, kk = k - 8*hi,
//   vgpr i = kk<8 ? kk/2 : 4+(kk-16)/2, element = 2i + (kk&1)

// ---------------------------------------------------------------------------
// One-time weight pre-swizzle: W1[512,64] -> W1f bf16 fragment order
// W1f[((kc*4 + wvt)*32 + lane)*16 + elem]
// ---------------------------------------------------------------------------
__global__ void k_prep_w1(const float* __restrict__ W1, __bf16* __restrict__ W1f)
{
    int idx = blockIdx.x * blockDim.x + threadIdx.x;
    if (idx >= 16 * 4 * 32 * 16) return;
    int elem = idx & 15;
    int lane = (idx >> 4) & 31;
    int wvt  = (idx >> 9) & 3;
    int kc   = idx >> 11;
    int i = elem >> 1, p = elem & 1;
    int hi = lane >> 4, nn = lane & 15;
    int k = kc * 32 + ((elem >> 3) << 4) + hi * 8 + (elem & 7);
    W1f[idx] = f2bf(W1[k * HID + wvt * 16 + nn]);
}

// W2[128,16] -> W2f bf16 fragment order: W2f[(kc*32 + lane)*16 + elem]
__global__ void k_prep_w2(const float* __restrict__ W2, __bf16* __restrict__ W2f)
{
    int idx = blockIdx.x * blockDim.x + threadIdx.x;
    if (idx >= 4 * 32 * 16) return;
    int elem = idx & 15;
    int lane = (idx >> 4) & 31;
    int kc   = idx >> 9;
    int i = elem >> 1, p = elem & 1;
    int hi = lane >> 4, nn = lane & 15;
    int k = kc * 32 + ((elem >> 3) << 4) + hi * 8 + (elem & 7);
    W2f[idx] = f2bf(W2[k * NCLS + nn]);
}

// ---------------------------------------------------------------------------
// GEMM1: xw[n,64] = x[n,512] @ W1[512,64]  (bf16 WMMA, fp32 accum)
// Block = 32 rows (2 row-tiles), 4 waves; wave w owns N-tile [16w,16w+16).
// A staged in LDS in fragment order (double buffered, 1 barrier/chunk);
// B operands loaded fragment-ordered from global (L2 resident).
// ---------------------------------------------------------------------------
__launch_bounds__(128)
__global__ void k_gemm1(const float* __restrict__ x, const __bf16* __restrict__ W1f,
                        float* __restrict__ xw, int n)
{
    __shared__ __attribute__((aligned(32))) __bf16 Afrag[2][2][32][16];
    const int t    = threadIdx.x;
    const int lane = t & 31;
    const int wv   = t >> 5;
    const int m    = lane & 15;
    const int hi   = lane >> 4;
    const int row0 = blockIdx.x * 32;

    v8f acc0 = {}, acc1 = {};
    for (int kc = 0; kc < F_IN / 32; ++kc) {
        const int K0 = kc * 32;
        const int buf = kc & 1;
        // stage A: 32 rows x 32 k as bf16 fragments; 2 groups of 4 per thread
#pragma unroll
        for (int gg = 0; gg < 2; ++gg) {
            int g  = t + gg * 128;          // 0..255
            int r  = g >> 3;                // 0..31
            int cg = (g & 7) * 4;           // 0,4,...,28
            int rt = r >> 4, mm = r & 15;
            int gr = row0 + r; if (gr >= n) gr = n - 1;
            float4 v = *reinterpret_cast<const float4*>(x + (long long)gr * F_IN + K0 + cg);
            int h  = (cg >> 3) & 1;
            int kk = cg - 8 * h;
            int iv = (kk < 8) ? (kk >> 1) : (4 + ((kk - 16) >> 1));
            v4bf q; q[0] = f2bf(v.x); q[1] = f2bf(v.y); q[2] = f2bf(v.z); q[3] = f2bf(v.w);
            *reinterpret_cast<v4bf*>(&Afrag[buf][rt][mm + 16 * h][2 * iv]) = q;
        }
        __syncthreads();
        // operands: contiguous 32B per lane
        v16bf a0 = *reinterpret_cast<const v16bf*>(&Afrag[buf][0][lane][0]);
        v16bf a1 = *reinterpret_cast<const v16bf*>(&Afrag[buf][1][lane][0]);
        v16bf b  = *reinterpret_cast<const v16bf*>(W1f + (((kc * 4 + wv) * 32 + lane) << 4));
        acc0 = __builtin_amdgcn_wmma_f32_16x16x32_bf16(false, a0, false, b,
                                                       (short)0, acc0, false, false);
        acc1 = __builtin_amdgcn_wmma_f32_16x16x32_bf16(false, a1, false, b,
                                                       (short)0, acc1, false, false);
        // no trailing barrier: double buffered
    }
#pragma unroll
    for (int r = 0; r < 8; ++r) {
        int mo = r + hi * 8;
        int rowA = row0 + mo, rowB = row0 + 16 + mo;
        if (rowA < n) xw[(long long)rowA * HID + wv * 16 + m] = acc0[r];
        if (rowB < n) xw[(long long)rowB * HID + wv * 16 + m] = acc1[r];
    }
}

// ---------------------------------------------------------------------------
// GEMM2: r1w[n,16] = relu(h1[n,128]) @ W2[128,16]
// Block = 64 rows, wave w owns rows [16w,16w+16); single N tile.
// ---------------------------------------------------------------------------
__launch_bounds__(128)
__global__ void k_gemm2(const float* __restrict__ h1, const __bf16* __restrict__ W2f,
                        float* __restrict__ r1w, int n)
{
    __shared__ __attribute__((aligned(32))) __bf16 Afrag[2][4][32][16];
    const int t    = threadIdx.x;
    const int lane = t & 31;
    const int wv   = t >> 5;
    const int m    = lane & 15;
    const int hi   = lane >> 4;
    const int row0 = blockIdx.x * 64;

    v8f acc = {};
#pragma unroll
    for (int kc = 0; kc < 4; ++kc) {
        const int K0 = kc * 32;
        const int buf = kc & 1;
        // stage A: 64 rows x 32 k, ReLU fused; 4 groups of 4 per thread
#pragma unroll
        for (int gg = 0; gg < 4; ++gg) {
            int g  = t + gg * 128;          // 0..511
            int r  = g >> 3;                // 0..63
            int cg = (g & 7) * 4;
            int rt = r >> 4, mm = r & 15;
            int gr = row0 + r; if (gr >= n) gr = n - 1;
            float4 v = *reinterpret_cast<const float4*>(h1 + (long long)gr * 128 + K0 + cg);
            int h  = (cg >> 3) & 1;
            int kk = cg - 8 * h;
            int iv = (kk < 8) ? (kk >> 1) : (4 + ((kk - 16) >> 1));
            v4bf q;
            q[0] = f2bf(fmaxf(v.x, 0.f)); q[1] = f2bf(fmaxf(v.y, 0.f));
            q[2] = f2bf(fmaxf(v.z, 0.f)); q[3] = f2bf(fmaxf(v.w, 0.f));
            *reinterpret_cast<v4bf*>(&Afrag[buf][rt][mm + 16 * h][2 * iv]) = q;
        }
        __syncthreads();
        v16bf a = *reinterpret_cast<const v16bf*>(&Afrag[buf][wv][lane][0]);
        v16bf b = *reinterpret_cast<const v16bf*>(W2f + ((kc * 32 + lane) << 4));
        acc = __builtin_amdgcn_wmma_f32_16x16x32_bf16(false, a, false, b,
                                                      (short)0, acc, false, false);
    }
#pragma unroll
    for (int r = 0; r < 8; ++r) {
        int row = row0 + wv * 16 + r + hi * 8;
        if (row < n) r1w[(long long)row * NCLS + m] = acc[r];
    }
}

// ---------------------------------------------------------------------------
// Sparse-aggregation helpers
// ---------------------------------------------------------------------------
__global__ void k_zero(float* p, long long n)
{
    long long i = blockIdx.x * (long long)blockDim.x + threadIdx.x;
    if (i < n) p[i] = 0.f;
}

__global__ void k_count(const int* __restrict__ tt, long long E, float* deg)
{
    long long e = blockIdx.x * (long long)blockDim.x + threadIdx.x;
    if (e < E) atomicAdd(&deg[tt[e]], 1.0f);
}

__global__ void k_dinv(float* d, int n)    // deg -> 1/sqrt(deg + selfloop)
{
    int i = blockIdx.x * blockDim.x + threadIdx.x;
    if (i < n) d[i] = rsqrtf(d[i] + 1.0f);
}

// self-loop contribution + bias for layer 1 (writes every element of h1)
__global__ void k_init_h1(const float* __restrict__ xw, const float* __restrict__ d1,
                          const float* __restrict__ d2, const float* __restrict__ b1,
                          float* __restrict__ h1, int n)
{
    int gid = blockIdx.x * blockDim.x + threadIdx.x;
    if (gid >= n * HID) return;
    int node = gid >> 6, c = gid & 63;
    float v = xw[(long long)node * HID + c], bb = b1[c];
    float a1 = d1[node], a2 = d2[node];
    h1[(long long)node * 128 + c]       = v * a1 * a1 + bb;
    h1[(long long)node * 128 + HID + c] = v * a2 * a2 + bb;
}

__global__ void k_init_h2(const float* __restrict__ r1w, const float* __restrict__ d1,
                          const float* __restrict__ d2, const float* __restrict__ b2,
                          float* __restrict__ h2, int n)
{
    int gid = blockIdx.x * blockDim.x + threadIdx.x;
    if (gid >= n * NCLS) return;
    int node = gid >> 4, c = gid & 15;
    float v = r1w[(long long)node * NCLS + c], bb = b2[c];
    float a1 = d1[node], a2 = d2[node];
    h2[(long long)node * 32 + c]        = v * a1 * a1 + bb;
    h2[(long long)node * 32 + NCLS + c] = v * a2 * a2 + bb;
}

// edge scatter: out[t, col0+4q..] += feat[s, 4q..] * dinv[s]*dinv[t]
__global__ void k_scatter(const float* __restrict__ feat, int fdim,
                          const int* __restrict__ ss, const int* __restrict__ tt,
                          long long E, const float* __restrict__ dinv,
                          float* __restrict__ out, int odim, int col0, int qlog)
{
    long long gid = blockIdx.x * (long long)blockDim.x + threadIdx.x;
    if (gid >= (E << qlog)) return;
    long long e = gid >> qlog;
    int q = (int)(gid & ((1 << qlog) - 1));
    int s = ss[e], tn = tt[e];
    float nrm = dinv[s] * dinv[tn];
    const float4 v = *reinterpret_cast<const float4*>(feat + (long long)s * fdim + q * 4);
    float* o = out + (long long)tn * odim + col0 + q * 4;
    atomicAdd(o + 0, v.x * nrm);
    atomicAdd(o + 1, v.y * nrm);
    atomicAdd(o + 2, v.z * nrm);
    atomicAdd(o + 3, v.w * nrm);
}

// final: out = log_softmax(h2[n,32] @ Wlin[32,16] + blin)
__launch_bounds__(256)
__global__ void k_final(const float* __restrict__ h2, const float* __restrict__ Wlin,
                        const float* __restrict__ blin, float* __restrict__ out, int n)
{
    __shared__ float Ws[32 * 16];
    __shared__ float bs[16];
    for (int i = threadIdx.x; i < 32 * 16; i += blockDim.x) Ws[i] = Wlin[i];
    for (int i = threadIdx.x; i < 16; i += blockDim.x) bs[i] = blin[i];
    __syncthreads();
    int node = blockIdx.x * blockDim.x + threadIdx.x;
    if (node >= n) return;
    float r[32];
#pragma unroll
    for (int i = 0; i < 32; ++i) r[i] = h2[(long long)node * 32 + i];
    float f[16]; float mx = -1e30f;
#pragma unroll
    for (int c = 0; c < 16; ++c) {
        float s = bs[c];
#pragma unroll
        for (int i = 0; i < 32; ++i) s += r[i] * Ws[i * 16 + c];
        f[c] = s; mx = fmaxf(mx, s);
    }
    float se = 0.f;
#pragma unroll
    for (int c = 0; c < 16; ++c) se += expf(f[c] - mx);
    float lse = mx + logf(se);
#pragma unroll
    for (int c = 0; c < 16; ++c) out[(long long)node * 16 + c] = f[c] - lse;
}

// ---------------------------------------------------------------------------
extern "C" void kernel_launch(void* const* d_in, const int* in_sizes, int n_in,
                              void* d_out, int out_size, void* d_ws, size_t ws_size,
                              hipStream_t stream)
{
    (void)n_in; (void)out_size; (void)ws_size;
    const float* x  = (const float*)d_in[0];
    const int*   e1 = (const int*)d_in[1];
    const int*   e2 = (const int*)d_in[2];
    const float* W1 = (const float*)d_in[3];
    const float* b1 = (const float*)d_in[4];
    const float* W2 = (const float*)d_in[5];
    const float* b2 = (const float*)d_in[6];
    const float* Wl = (const float*)d_in[7];
    const float* bl = (const float*)d_in[8];
    float* out = (float*)d_out;

    const int n = in_sizes[0] / F_IN;
    const long long E1 = in_sizes[1] / 2;
    const long long E2 = in_sizes[2] / 2;
    const int *s1 = e1, *t1 = e1 + E1;
    const int *s2 = e2, *t2 = e2 + E2;

    char* p = (char*)d_ws;
    auto carve = [&](size_t bytes) -> void* {
        void* r = (void*)p; p += (bytes + 255) & ~(size_t)255; return r;
    };
    float*  xw    = (float*)carve((size_t)n * HID * 4);
    float*  r1w   = (float*)carve((size_t)n * NCLS * 4);
    float*  h1    = (float*)carve((size_t)n * 2 * HID * 4);
    float*  h2    = (float*)carve((size_t)n * 2 * NCLS * 4);
    float*  dinv1 = (float*)carve((size_t)n * 4);
    float*  dinv2 = (float*)carve((size_t)n * 4);
    __bf16* W1f   = (__bf16*)carve((size_t)F_IN * HID * 2);   // 64 KB
    __bf16* W2f   = (__bf16*)carve((size_t)128 * NCLS * 2);   // 4 KB

    const int B = 256;
    // weight pre-swizzle into WMMA fragment order (bf16)
    k_prep_w1<<<(16 * 4 * 32 * 16 + B - 1) / B, B, 0, stream>>>(W1, W1f);
    k_prep_w2<<<(4 * 32 * 16 + B - 1) / B, B, 0, stream>>>(W2, W2f);

    // symmetric-normalization coefficients (deg + self loop)
    k_zero<<<(n + B - 1) / B, B, 0, stream>>>(dinv1, n);
    k_zero<<<(n + B - 1) / B, B, 0, stream>>>(dinv2, n);
    k_count<<<(int)((E1 + B - 1) / B), B, 0, stream>>>(t1, E1, dinv1);
    k_count<<<(int)((E2 + B - 1) / B), B, 0, stream>>>(t2, E2, dinv2);
    k_dinv<<<(n + B - 1) / B, B, 0, stream>>>(dinv1, n);
    k_dinv<<<(n + B - 1) / B, B, 0, stream>>>(dinv2, n);

    // layer 1: dense WMMA GEMM shared by both edge sets, then scatter
    k_gemm1<<<(n + 31) / 32, 128, 0, stream>>>(x, W1f, xw, n);
    k_init_h1<<<(n * HID + B - 1) / B, B, 0, stream>>>(xw, dinv1, dinv2, b1, h1, n);
    k_scatter<<<(int)(((E1 << 4) + B - 1) / B), B, 0, stream>>>(xw, HID, s1, t1, E1, dinv1, h1, 2 * HID, 0, 4);
    k_scatter<<<(int)(((E2 << 4) + B - 1) / B), B, 0, stream>>>(xw, HID, s2, t2, E2, dinv2, h1, 2 * HID, HID, 4);

    // layer 2
    k_gemm2<<<(n + 63) / 64, 128, 0, stream>>>(h1, W2f, r1w, n);
    k_init_h2<<<(n * NCLS + B - 1) / B, B, 0, stream>>>(r1w, dinv1, dinv2, b2, h2, n);
    k_scatter<<<(int)(((E1 << 2) + B - 1) / B), B, 0, stream>>>(r1w, NCLS, s1, t1, E1, dinv1, h2, 2 * NCLS, 0, 2);
    k_scatter<<<(int)(((E2 << 2) + B - 1) / B), B, 0, stream>>>(r1w, NCLS, s2, t2, E2, dinv2, h2, 2 * NCLS, NCLS, 2);

    // final linear + log_softmax
    k_final<<<(n + B - 1) / B, B, 0, stream>>>(h2, Wl, bl, out, n);
}
